// MultiHeadAttentionModule_82411832476080
// MI455X (gfx1250) — matrix operand order repaired and checked by reading, not compiled, not run on hardware
//
#include <hip/hip_runtime.h>
#include <hip/hip_bf16.h>

// ---------------------------------------------------------------------------
// MultiHeadAttention forward for MI455X (gfx1250, wave32, WMMA + TDM).
//   B=2, S=2048, D=1024, H=16, HD=64
// Kernel 1: fused QKV projection, register-blocked 32x64 per wave
//           (8 x v_wmma_f32_16x16x32_bf16 per K-step).
// Kernel 2: causal flash attention; 128 q-rows per block; K/V chunks staged
//           once per block into LDS by the Tensor Data Mover and shared by
//           4 waves; WMMA for Q@K^T and P@V; online softmax via shuffles.
// ---------------------------------------------------------------------------

#define B_  2
#define S_  2048
#define D_  1024
#define H_  16
#define HD_ 64

typedef __attribute__((ext_vector_type(16))) __bf16 v16bf;
typedef __attribute__((ext_vector_type(8)))  __bf16 v8bf;
typedef __attribute__((ext_vector_type(8)))  float  v8f;

typedef unsigned int uint_t;
typedef __attribute__((ext_vector_type(4))) uint_t uint32x4;
typedef __attribute__((ext_vector_type(8))) int    int32x8;
typedef __attribute__((ext_vector_type(4))) int    int32x4;

// K-offset base for 16-bit A-matrix fragment (16x32), contiguous order:
// lanes 0-15: vgpr v holds K = 2v, +1 ; lanes 16-31: +16.
__device__ __forceinline__ int a_kbase(int v, int half) {
    return 2 * v + (half ? 16 : 0);
}

__device__ __forceinline__ v8f vzero8() {
    return (v8f){0.f, 0.f, 0.f, 0.f, 0.f, 0.f, 0.f, 0.f};
}

// ---------------------------------------------------------------------------
// TDM: issue a 2-D tensor tile load (bf16 elements) from global to LDS.
// Descriptor layout per CDNA5 ISA 8.3/8.4 (D# group0 128b, group1 256b).
// ---------------------------------------------------------------------------
__device__ __forceinline__ void tdm_load_2d_bf16(unsigned lds_addr,
                                                 const void* gaddr,
                                                 unsigned tile_w, unsigned tile_h,
                                                 unsigned tensor_w, unsigned tensor_h,
                                                 unsigned stride_elems)
{
    const unsigned long long ga = (unsigned long long)(uintptr_t)gaddr;

    uint32x4 g0;
    g0[0] = 1u;                                      // count=1, user D#
    g0[1] = lds_addr;                                // LDS byte address
    g0[2] = (uint_t)(ga & 0xffffffffull);            // global_addr[31:0]
    g0[3] = (uint_t)((ga >> 32) & 0x01ffffffull)     // global_addr[56:32]
          | (2u << 30);                              // type = 2 ("image")

    int32x8 g1;
    g1[0] = (int)(1u << 16);                         // data_size=1 (2B); mask=0
    g1[1] = (int)((tensor_w & 0xffffu) << 16);       // tensor_dim0[15:0] @ bit48
    g1[2] = (int)(((tensor_w >> 16) & 0xffffu)
          | ((tensor_h & 0xffffu) << 16));           // dim0 hi | dim1 lo
    g1[3] = (int)(((tensor_h >> 16) & 0xffffu)
          | ((tile_w & 0xffffu) << 16));             // dim1 hi | tile_dim0
    g1[4] = (int)(tile_h & 0xffffu);                 // tile_dim1 (tile_dim2=0)
    g1[5] = (int)stride_elems;                       // tensor_dim0_stride[31:0]
    g1[6] = 0;
    g1[7] = 0;

    const int32x4 z4 = {0, 0, 0, 0};
#if __clang_major__ >= 23
    const int32x8 z8 = {0, 0, 0, 0, 0, 0, 0, 0};
    __builtin_amdgcn_tensor_load_to_lds(g0, g1, z4, z4, z8, 0);
#else
    __builtin_amdgcn_tensor_load_to_lds(g0, g1, z4, z4, 0);
#endif
}

// ---------------------------------------------------------------------------
// Kernel 1: fused QKV projection.  y[e] = sum_d x[d] * W[e,d]  (Linear, no bias)
// grid = (B*S/128, D/64, 3), block = 128.  Wave w computes a 32x64 tile:
// rows [row0+32w, +32), cols [e0, e0+64): 2 A-frags x 4 B-frags -> 8 WMMA/K-step.
// Output layout: [B, H, S, HD] bf16.
// ---------------------------------------------------------------------------
__global__ __launch_bounds__(128)
void qkv_proj_kernel(const float* __restrict__ Q, const float* __restrict__ K,
                     const float* __restrict__ V,
                     const float* __restrict__ Wq, const float* __restrict__ Wk,
                     const float* __restrict__ Wv,
                     __bf16* __restrict__ Qb, __bf16* __restrict__ Kb,
                     __bf16* __restrict__ Vb)
{
    const int lane = threadIdx.x & 31;
    const int wave = threadIdx.x >> 5;
    const int half = lane >> 4;
    const int l16  = lane & 15;

    const int z    = blockIdx.z;                        // 0:q 1:k 2:v
    const int row0 = blockIdx.x * 128 + wave * 32;      // global row in [0, B*S)
    const int e0   = blockIdx.y * 64;                   // output feature tile

    const float*  X = (z == 0) ? Q : (z == 1) ? K : V;
    const float*  W = (z == 0) ? Wq : (z == 1) ? Wk : Wv;
    __bf16*       Y = (z == 0) ? Qb : (z == 1) ? Kb : Vb;

    const float* xrow[2];
    xrow[0] = X + (size_t)(row0 + l16) * D_;
    xrow[1] = X + (size_t)(row0 + 16 + l16) * D_;

    const float* wrow[4];
#pragma unroll
    for (int t = 0; t < 4; ++t)
        wrow[t] = W + (size_t)(e0 + t * 16 + l16) * D_ + 16 * half;

    v8f c[2][4];
#pragma unroll
    for (int rt = 0; rt < 2; ++rt)
#pragma unroll
        for (int t = 0; t < 4; ++t) c[rt][t] = vzero8();

    for (int kk = 0; kk < D_; kk += 32) {
        // A fragments: x rows (two 16 x 32 tiles)
        v16bf a[2];
#pragma unroll
        for (int rt = 0; rt < 2; ++rt) {
#pragma unroll
            for (int v = 0; v < 8; ++v) {
                const int kb = a_kbase(v, half);
                a[rt][2 * v + 0] = (__bf16)xrow[rt][kk + kb + 0];
                a[rt][2 * v + 1] = (__bf16)xrow[rt][kk + kb + 1];
            }
        }
        // B fragments: W^T chunks (32 x 16); lane column n=l16, K=2v+p+16*half
#pragma unroll
        for (int t = 0; t < 4; ++t) {
            v16bf bf;
#pragma unroll
            for (int v = 0; v < 8; ++v) {
                bf[2 * v + 0] = (__bf16)wrow[t][kk + 2 * v + 0];
                bf[2 * v + 1] = (__bf16)wrow[t][kk + 2 * v + 1];
            }
#pragma unroll
            for (int rt = 0; rt < 2; ++rt)
                c[rt][t] = __builtin_amdgcn_wmma_f32_16x16x32_bf16(
                               false, a[rt], false, bf, (short)0, c[rt][t],
                               false, false);
        }
    }

    // C layout: vgpr r, lane -> row = r + 8*half, col = l16
#pragma unroll
    for (int t = 0; t < 4; ++t) {
        const int e  = e0 + t * 16 + l16;
        const int h  = e / HD_;
        const int hd = e % HD_;
#pragma unroll
        for (int rt = 0; rt < 2; ++rt) {
#pragma unroll
            for (int r = 0; r < 8; ++r) {
                const int m = row0 + rt * 16 + r + 8 * half;
                const int b = m / S_;
                const int s = m % S_;
                Y[((size_t)(b * H_ + h) * S_ + s) * HD_ + hd] = (__bf16)c[rt][t][r];
            }
        }
    }
}

// ---------------------------------------------------------------------------
// Kernel 2: causal flash attention.
// grid = (B*H, S/128), block = 128 (4 waves).  Each block owns 128 query rows;
// wave w owns rows [q0, q0+32) as two 16-row WMMA tiles.  K/V chunks (32x64
// bf16) are TDM-staged into LDS once per block and shared by all waves.
// ---------------------------------------------------------------------------
__global__ __launch_bounds__(128)
void attn_kernel(const __bf16* __restrict__ Qb, const __bf16* __restrict__ Kb,
                 const __bf16* __restrict__ Vb, float* __restrict__ O)
{
    __shared__ __align__(16) __bf16 kbuf[32 * HD_];      // K chunk
    __shared__ __align__(16) __bf16 vbuf[32 * HD_];      // V chunk
    __shared__ __align__(16) __bf16 pbuf[4][2][16 * 32]; // P staging / wave,qt

    const int lane = threadIdx.x & 31;
    const int wave = threadIdx.x >> 5;
    const int half = lane >> 4;
    const int l16  = lane & 15;

    const int bh   = blockIdx.x;             // b*H + h
    const int qblk = blockIdx.y * 128;
    const int q0   = qblk + wave * 32;       // this wave's first query row
    const int b    = bh / H_;
    const int h    = bh % H_;

    const __bf16* qbase = Qb + (size_t)bh * S_ * HD_;
    const __bf16* kbase = Kb + (size_t)bh * S_ * HD_;
    const __bf16* vbase = Vb + (size_t)bh * S_ * HD_;

    const unsigned lds_k = (unsigned)(uintptr_t)&kbuf[0];
    const unsigned lds_v = (unsigned)(uintptr_t)&vbuf[0];

    // ---- load Q tiles (2 x 16 x 64) as K=32 A-fragments ----
    v16bf qa[2][2];
#pragma unroll
    for (int qt = 0; qt < 2; ++qt) {
        const __bf16* qrow = qbase + (size_t)(q0 + qt * 16 + l16) * HD_;
#pragma unroll
        for (int cfrag = 0; cfrag < 2; ++cfrag) {
#pragma unroll
            for (int v = 0; v < 8; ++v) {
                const int kb = a_kbase(v, half) + cfrag * 32;
                qa[qt][cfrag][2 * v + 0] = qrow[kb + 0];
                qa[qt][cfrag][2 * v + 1] = qrow[kb + 1];
            }
        }
    }

    v8f acc[2][4];
    float mrow[2][8], lrow[2][8];
#pragma unroll
    for (int qt = 0; qt < 2; ++qt) {
#pragma unroll
        for (int t = 0; t < 4; ++t) acc[qt][t] = vzero8();
#pragma unroll
        for (int r = 0; r < 8; ++r) { mrow[qt][r] = -3.0e38f; lrow[qt][r] = 0.f; }
    }

    const float sc = 0.125f;   // 1/sqrt(HD)
    const int kmax = qblk + 128;

    for (int k0 = 0; k0 < kmax; k0 += 32) {
        // All waves' LDS reads of the previous chunk must drain before the
        // TDM rewrites kbuf/vbuf (TDM is unordered vs DS ops).
        asm volatile("s_wait_dscnt 0" ::: "memory");
        __syncthreads();

        if (wave == 0) {
            tdm_load_2d_bf16(lds_k, kbase + (size_t)k0 * HD_,
                             /*tile*/ HD_, 32, /*tensor*/ HD_, S_, HD_);
            tdm_load_2d_bf16(lds_v, vbase + (size_t)k0 * HD_,
                             /*tile*/ HD_, 32, /*tensor*/ HD_, S_, HD_);
            __builtin_prefetch(kbase + (size_t)(k0 + 32) * HD_, 0, 1);
            __builtin_prefetch(vbase + (size_t)(k0 + 32) * HD_, 0, 1);
            __builtin_amdgcn_s_wait_tensorcnt(0);
        }
        __syncthreads();

        // Causal: this wave only consumes chunks that intersect keys <= rows.
        if (k0 < q0 + 32) {
            // ---- scores: per q-tile, two 16x16 key tiles ----
            v8f sacc[2][2];
#pragma unroll
            for (int nt = 0; nt < 2; ++nt) {
                // B fragment (shared by both q-tiles): lane col = key, K = hd.
                const __bf16* krow = kbuf + (nt * 16 + l16) * HD_ + 16 * half;
                const v16bf kf0 = *(const v16bf*)(krow);
                const v16bf kf1 = *(const v16bf*)(krow + 32);
#pragma unroll
                for (int qt = 0; qt < 2; ++qt) {
                    v8f s = __builtin_amdgcn_wmma_f32_16x16x32_bf16(
                                false, qa[qt][0], false, kf0, (short)0,
                                vzero8(), false, false);
                    s = __builtin_amdgcn_wmma_f32_16x16x32_bf16(
                                false, qa[qt][1], false, kf1, (short)0, s,
                                false, false);
                    sacc[qt][nt] = s;
                }
            }

            // ---- scale + causal mask + online softmax ----
#pragma unroll
            for (int qt = 0; qt < 2; ++qt) {
#pragma unroll
                for (int nt = 0; nt < 2; ++nt) {
#pragma unroll
                    for (int r = 0; r < 8; ++r) {
                        const int col = k0 + nt * 16 + l16;
                        const int row = q0 + qt * 16 + r + 8 * half;
                        float sv = sacc[qt][nt][r] * sc;
                        if (col > row) sv = -3.0e38f;
                        sacc[qt][nt][r] = sv;
                    }
                }
#pragma unroll
                for (int r = 0; r < 8; ++r) {
                    float mx = fmaxf(sacc[qt][0][r], sacc[qt][1][r]);
#pragma unroll
                    for (int o = 1; o < 16; o <<= 1)
                        mx = fmaxf(mx, __shfl_xor(mx, o, 16));
                    const float mn = fmaxf(mrow[qt][r], mx);
                    const float al = __expf(mrow[qt][r] - mn);
                    mrow[qt][r] = mn;
                    const float p0 = __expf(sacc[qt][0][r] - mn);
                    const float p1 = __expf(sacc[qt][1][r] - mn);
                    sacc[qt][0][r] = p0;
                    sacc[qt][1][r] = p1;
                    float rs = p0 + p1;
#pragma unroll
                    for (int o = 1; o < 16; o <<= 1)
                        rs += __shfl_xor(rs, o, 16);
                    lrow[qt][r] = lrow[qt][r] * al + rs;
#pragma unroll
                    for (int t = 0; t < 4; ++t)
                        acc[qt][t][r] *= al;
                }
            }

            // ---- C-layout -> A-layout for P via per-wave LDS staging ----
#pragma unroll
            for (int qt = 0; qt < 2; ++qt)
#pragma unroll
                for (int nt = 0; nt < 2; ++nt)
#pragma unroll
                    for (int r = 0; r < 8; ++r)
                        pbuf[wave][qt][(r + 8 * half) * 32 + nt * 16 + l16] =
                            (__bf16)sacc[qt][nt][r];

            asm volatile("s_wait_dscnt 0" ::: "memory");  // wave-local RAW

            v16bf pa[2];
#pragma unroll
            for (int qt = 0; qt < 2; ++qt) {
                const __bf16* prow = &pbuf[wave][qt][l16 * 32];
                const v8bf lo = *(const v8bf*)(prow + 16 * half);
                const v8bf hi = *(const v8bf*)(prow + 16 * half + 8);
#pragma unroll
                for (int i = 0; i < 8; ++i) {
                    pa[qt][i] = lo[i];
                    pa[qt][8 + i] = hi[i];
                }
            }

            // ---- ctx += P (16x32) @ V (32x64); V-fragment shared by q-tiles ----
#pragma unroll
            for (int t = 0; t < 4; ++t) {
                const __bf16* vcol = vbuf + t * 16 + l16;
                v16bf vf;
#pragma unroll
                for (int v = 0; v < 8; ++v) {
                    const int kd = 2 * v + 16 * half;
                    vf[2 * v + 0] = vcol[(kd + 0) * HD_];
                    vf[2 * v + 1] = vcol[(kd + 1) * HD_];
                }
#pragma unroll
                for (int qt = 0; qt < 2; ++qt)
                    acc[qt][t] = __builtin_amdgcn_wmma_f32_16x16x32_bf16(
                                     false, pa[qt], false, vf, (short)0,
                                     acc[qt][t], false, false);
            }
        }
    }

    // ---- normalize and store (out is fp32 [B, S, D]) ----
#pragma unroll
    for (int qt = 0; qt < 2; ++qt) {
        float inv[8];
#pragma unroll
        for (int r = 0; r < 8; ++r) inv[r] = 1.0f / lrow[qt][r];
#pragma unroll
        for (int t = 0; t < 4; ++t) {
#pragma unroll
            for (int r = 0; r < 8; ++r) {
                const int s = q0 + qt * 16 + r + 8 * half;
                O[((size_t)b * S_ + s) * D_ + h * HD_ + t * 16 + l16] =
                    acc[qt][t][r] * inv[r];
            }
        }
    }
}

// ---------------------------------------------------------------------------
// Host-side launcher.
// Workspace: q/k/v bf16 [B,H,S,HD] = 3 * 8 MB = 24 MB.
// ---------------------------------------------------------------------------
extern "C" void kernel_launch(void* const* d_in, const int* in_sizes, int n_in,
                              void* d_out, int out_size, void* d_ws, size_t ws_size,
                              hipStream_t stream) {
    const float* query = (const float*)d_in[0];
    const float* key   = (const float*)d_in[1];
    const float* value = (const float*)d_in[2];
    const float* Wq    = (const float*)d_in[3];
    const float* Wk    = (const float*)d_in[4];
    const float* Wv    = (const float*)d_in[5];
    float* out = (float*)d_out;

    const size_t N = (size_t)B_ * S_ * D_;
    __bf16* qb = (__bf16*)d_ws;
    __bf16* kb = qb + N;
    __bf16* vb = kb + N;

    dim3 g1(B_ * S_ / 128, D_ / 64, 3);
    qkv_proj_kernel<<<g1, 128, 0, stream>>>(query, key, value, Wq, Wk, Wv,
                                            qb, kb, vb);

    dim3 g2(B_ * H_, S_ / 128);
    attn_kernel<<<g2, 128, 0, stream>>>(qb, kb, vb, out);
}
